// Classifier_74517682585655
// MI455X (gfx1250) — matrix operand order repaired and checked by reading, not compiled, not run on hardware
//
#include <hip/hip_runtime.h>

typedef __attribute__((ext_vector_type(16))) __bf16 v16bf;
typedef __attribute__((ext_vector_type(8)))  __bf16 v8bf;
typedef __attribute__((ext_vector_type(4)))  __bf16 v4bf;
typedef __attribute__((ext_vector_type(8)))  float  v8f;

// ---------------- Layer-1 GEMM: [8192,6400] x [6400,256] + b1, relu ----------------
// Block tile 128(M) x 128(N), K-tile 64. 256 threads = 8 waves (4 M x 2 N).
// Each wave: 32x64 tile -> 2x4 accumulators of v_wmma_f32_16x16x32_bf16.
// Double-buffered LDS pipeline; A staged row-major, B staged transposed [n][k]
// with 4x4 per-thread patches so converts are packed (v_cvt_pk_bf16_f32) and
// stores are ds_store_b64 along the contiguous k axis.

static constexpr int MTOT = 8192;
static constexpr int KTOT = 6400;
static constexpr int NTOT = 256;
static constexpr int BM   = 128;
static constexpr int BN   = 128;
static constexpr int BK   = 64;
static constexpr int NT   = KTOT / BK;  // 100 K-tiles
static constexpr int LDA  = BK + 8;     // 144B row stride -> conflict-free frag reads
static constexpr int LDBT = BK + 8;     // transposed B row stride (per n), 16B-aligned

// fragment = [p .. p+7] ++ [p+gap .. p+gap+7]  (bf16 elements)
static __device__ __forceinline__ v16bf load_frag(const __bf16* p, int gap) {
  v8bf lo = *(const v8bf*)(p);
  v8bf hi = *(const v8bf*)(p + gap);
  return __builtin_shufflevector(lo, hi, 0, 1, 2, 3, 4, 5, 6, 7,
                                         8, 9, 10, 11, 12, 13, 14, 15);
}

__global__ __launch_bounds__(256) void k_gemm1_relu(const float* __restrict__ X,
                                                    const float* __restrict__ W1,
                                                    const float* __restrict__ b1,
                                                    float* __restrict__ H1) {
  __shared__ __bf16 As[2][BM * LDA];   // 2 x 18 KB
  __shared__ __bf16 Bs[2][BN * LDBT];  // 2 x 18 KB (transposed: [n][k])

  const int tid  = threadIdx.x;
  const int lane = tid & 31;
  const int wid  = tid >> 5;
  const int half = lane >> 4;      // 0: lanes 0-15, 1: lanes 16-31
  const int l16  = lane & 15;

  const int bM = blockIdx.x * BM;
  const int bN = blockIdx.y * BN;
  const int wM = (wid & 3) * 32;   // wave M offset inside block
  const int wN = (wid >> 2) * 64;  // wave N offset inside block

  // A staging: thread -> rows arow+16i, k-cols acol..acol+3 (8 float4s)
  const int arow = tid >> 4;            // 0..15
  const int acol = (tid & 15) << 2;
  // B staging: thread -> 2 patches of 4(k) x 4(n): k = brow4 + 32i + j, n = bcol4..+3
  const int brow4 = wid << 2;           // 0..28
  const int bcol4 = (tid & 31) << 2;    // 0..124

  const float* gA = X  + (size_t)(bM + arow) * KTOT + acol;        // + 16i*KTOT + k0
  const float* gB = W1 + (size_t)brow4 * NTOT + bN + bcol4;        // + (32i + j + k0)*NTOT

  float4 ra[8];
  float4 rb[2][4];

  auto load_global = [&](int k0) {
#pragma unroll
    for (int i = 0; i < 8; ++i) {
      const float* p = gA + (size_t)(16 * i) * KTOT + k0;
      ra[i] = *(const float4*)p;
      __builtin_prefetch(p + 2 * BK, 0, 0);   // speculative SCOPE_SYS, safe past end
    }
#pragma unroll
    for (int i = 0; i < 2; ++i)
#pragma unroll
      for (int j = 0; j < 4; ++j)
        rb[i][j] = *(const float4*)(gB + (size_t)(k0 + 32 * i + j) * NTOT);
  };

  auto store_lds = [&](int buf) {
#pragma unroll
    for (int i = 0; i < 8; ++i) {
      v4bf v;
      v.x = (__bf16)ra[i].x; v.y = (__bf16)ra[i].y;
      v.z = (__bf16)ra[i].z; v.w = (__bf16)ra[i].w;
      *(v4bf*)&As[buf][(arow + 16 * i) * LDA + acol] = v;
    }
#pragma unroll
    for (int i = 0; i < 2; ++i) {
      const float* f0 = (const float*)&rb[i][0];
      const float* f1 = (const float*)&rb[i][1];
      const float* f2 = (const float*)&rb[i][2];
      const float* f3 = (const float*)&rb[i][3];
#pragma unroll
      for (int n = 0; n < 4; ++n) {
        v4bf v;                                  // 4 consecutive k for column bcol4+n
        v.x = (__bf16)f0[n]; v.y = (__bf16)f1[n];
        v.z = (__bf16)f2[n]; v.w = (__bf16)f3[n];
        *(v4bf*)&Bs[buf][(bcol4 + n) * LDBT + brow4 + 32 * i] = v;
      }
    }
  };

  v8f acc[2][4];
#pragma unroll
  for (int i = 0; i < 2; ++i)
#pragma unroll
    for (int j = 0; j < 4; ++j)
      acc[i][j] = (v8f){0.f, 0.f, 0.f, 0.f, 0.f, 0.f, 0.f, 0.f};

  auto compute = [&](int buf) {
#pragma unroll
    for (int ks = 0; ks < BK; ks += 32) {
      v16bf afrag[2];
#pragma unroll
      for (int sm = 0; sm < 2; ++sm)
        afrag[sm] = load_frag(&As[buf][(wM + sm * 16 + l16) * LDA + ks + 8 * half], 16);
      v16bf bfrag[4];
#pragma unroll
      for (int sn = 0; sn < 4; ++sn)
        bfrag[sn] = load_frag(&Bs[buf][(wN + sn * 16 + l16) * LDBT + ks + 8 * half], 16);
#pragma unroll
      for (int sm = 0; sm < 2; ++sm)
#pragma unroll
        for (int sn = 0; sn < 4; ++sn)
          acc[sm][sn] = __builtin_amdgcn_wmma_f32_16x16x32_bf16(
              false, afrag[sm], false, bfrag[sn],
              (short)0, acc[sm][sn], false, false);
    }
  };

  // ---- software pipeline: prologue, branch-free steady state, peeled tail ----
  load_global(0);
  store_lds(0);
  for (int t = 0; t < NT - 1; ++t) {
    load_global((t + 1) * BK);       // in flight during compute
    __syncthreads();                 // buf[t&1] fully staged by all waves
    compute(t & 1);
    store_lds((t + 1) & 1);          // waits on globals, writes the other buffer
  }
  __syncthreads();
  compute((NT - 1) & 1);

  // ---- epilogue: +b1, relu, store f32 H1 ----
  // C/D layout: VGPR i -> M = i + 8*half, N = l16
#pragma unroll
  for (int sn = 0; sn < 4; ++sn) {
    int col = bN + wN + sn * 16 + l16;
    float bias = b1[col];
#pragma unroll
    for (int sm = 0; sm < 2; ++sm) {
#pragma unroll
      for (int i = 0; i < 8; ++i) {
        int row = bM + wM + sm * 16 + half * 8 + i;
        float v = acc[sm][sn][i] + bias;
        H1[(size_t)row * NTOT + col] = v > 0.f ? v : 0.f;
      }
    }
  }
}

// ---------------- Layers 2+3: [8192,256]x[256,64] relu, then x[64,21] ----------------
// ~1% of total FLOPs: one row per thread, W2 broadcast from LDS, f32 VALU.

__global__ __launch_bounds__(256) void k_tail(const float* __restrict__ H1,
                                              const float* __restrict__ W2,
                                              const float* __restrict__ b2,
                                              const float* __restrict__ W3,
                                              const float* __restrict__ b3,
                                              float* __restrict__ out) {
  __shared__ float W2s[256 * 64];   // 64 KB
  for (int i = threadIdx.x; i < 256 * 64; i += 256)
    W2s[i] = W2[i];
  __syncthreads();

  const int row = blockIdx.x * 256 + threadIdx.x;
  const float* h = H1 + (size_t)row * 256;

  float h2[64];
#pragma unroll
  for (int n = 0; n < 64; ++n) h2[n] = b2[n];

  for (int k = 0; k < 256; ++k) {
    float xv = h[k];
    const float* w = &W2s[k * 64];
#pragma unroll
    for (int n = 0; n < 64; ++n)
      h2[n] = fmaf(xv, w[n], h2[n]);
  }
#pragma unroll
  for (int n = 0; n < 64; ++n) h2[n] = h2[n] > 0.f ? h2[n] : 0.f;

  float* o = out + (size_t)row * 21;
  for (int c = 0; c < 21; ++c) {
    float s = b3[c];
#pragma unroll
    for (int n = 0; n < 64; ++n)
      s = fmaf(h2[n], W3[n * 21 + c], s);
    o[c] = s;
  }
}

extern "C" void kernel_launch(void* const* d_in, const int* in_sizes, int n_in,
                              void* d_out, int out_size, void* d_ws, size_t ws_size,
                              hipStream_t stream) {
  const float* x  = (const float*)d_in[0];
  const float* W1 = (const float*)d_in[1];
  const float* b1 = (const float*)d_in[2];
  const float* W2 = (const float*)d_in[3];
  const float* b2 = (const float*)d_in[4];
  const float* W3 = (const float*)d_in[5];
  const float* b3 = (const float*)d_in[6];
  float* outp = (float*)d_out;
  float* H1   = (float*)d_ws;   // 8192 x 256 f32 = 8 MB scratch

  dim3 g1(MTOT / BM, NTOT / BN);          // 64 x 2
  k_gemm1_relu<<<g1, 256, 0, stream>>>(x, W1, b1, H1);
  k_tail<<<MTOT / 256, 256, 0, stream>>>(H1, W2, b2, W3, b3, outp);
}
